// TransformerEncoderLayer_16844861735791
// MI455X (gfx1250) — matrix operand order repaired and checked by reading, not compiled, not run on hardware
//
#include <hip/hip_runtime.h>
#include <hip/hip_bf16.h>
#include <math.h>

// Transformer encoder layer (pre-LN), B=2,S=4096,D=768,H=12,DH=64,DFF=3072
// All matmuls via v_wmma_f32_16x16x32_f16 (f16 operands, fp32 accum).
// Weights pre-swizzled once into B-fragment order -> inner loops feed WMMA
// with 32B contiguous global/LDS vector loads (no scalar ds_load_u16 chains).
// NOTE: reference never applies wo/bo -> d_in[8],[9] unused.

#define B_   2
#define S_   4096
#define D_   768
#define H_   12
#define DH_  64
#define DFF_ 3072
#define NROW (B_ * S_)   // 8192
#define EPS_ 1e-6f

typedef __attribute__((ext_vector_type(16))) _Float16 v16h;
typedef __attribute__((ext_vector_type(8)))  float    v8f;

// ---------------- WMMA fragment helpers (contiguous k order: k = 16*(lane/16) + elem) ----------------

// A fragment: 16x32 f16, row-major source with leading dim `ld` (halves).
// lane<16: row=lane, K 0..15 in elems 0..15
// lane>=16: row=lane-16, K 16..31
__device__ __forceinline__ v16h load_a_frag(const _Float16* p, int ld, int lane) {
    const int row = lane & 15;
    const int k0  = (lane & 16) ? 16 : 0;
    const _Float16* r0 = p + row * ld + k0;
    v16h a;
#pragma unroll
    for (int j = 0; j < 8; ++j) a[j] = r0[j];
#pragma unroll
    for (int j = 0; j < 8; ++j) a[8 + j] = r0[8 + j];
    return a;
}

// Contiguous 32B vector load of a pre-swizzled fragment (32B aligned).
__device__ __forceinline__ v16h load_v16h(const _Float16* p) {
    return *reinterpret_cast<const v16h*>(p);
}

__device__ __forceinline__ v8f wmma16x16x32(v16h a, v16h b, v8f c) {
    return __builtin_amdgcn_wmma_f32_16x16x32_f16(false, a, false, b,
                                                  (short)0, c, false, false);
}

// B-fragment swizzle index for a [K,N] matrix, 32x16 fragments:
// frag(kblk=k/32, ntile=n/16); lane = 16*((k>>4)&1) + (n&15); elem = k&15.
// Linear: ((kblk*(N/16) + ntile)*32 + lane)*16 + elem
__device__ __forceinline__ int bswz_idx(int k, int n, int N) {
    return (((k >> 5) * (N >> 4) + (n >> 4)) << 9) +
           (((k & 16) + (n & 15)) << 4) + (k & 15);
}

// ---------------- one-time weight convert + swizzle (fp32 -> f16 frag order) -

__global__ __launch_bounds__(256)
void wswz_kernel(const float* __restrict__ W, _Float16* __restrict__ Wsw,
                 int K, int N) {
    const int i = blockIdx.x * 256 + threadIdx.x;
    if (i >= K * N) return;
    const int k = i / N, n = i % N;
    Wsw[bswz_idx(k, n, N)] = (_Float16)W[i];
}

// ---------------- LayerNorm (torch-style: unbiased std, /(sigma+eps)) -------

__global__ __launch_bounds__(256)
void ln_kernel(const float* __restrict__ x, const float* __restrict__ gamma,
               const float* __restrict__ beta, _Float16* __restrict__ outh) {
    const int row = blockIdx.x;
    const float* xr = x + (size_t)row * D_;
    float s = 0.f, ss = 0.f;
    for (int c = threadIdx.x; c < D_; c += 256) {
        float v = xr[c]; s += v; ss += v * v;
    }
    __shared__ float rs[256], rss[256];
    rs[threadIdx.x] = s; rss[threadIdx.x] = ss;
    __syncthreads();
    for (int off = 128; off > 0; off >>= 1) {
        if (threadIdx.x < (unsigned)off) {
            rs[threadIdx.x]  += rs[threadIdx.x + off];
            rss[threadIdx.x] += rss[threadIdx.x + off];
        }
        __syncthreads();
    }
    const float mu    = rs[0] / (float)D_;
    const float var   = (rss[0] - (float)D_ * mu * mu) / (float)(D_ - 1);
    const float sigma = sqrtf(fmaxf(var, 0.f));
    const float inv   = 1.f / (sigma + EPS_);
    _Float16* o = outh + (size_t)row * D_;
    for (int c = threadIdx.x; c < D_; c += 256)
        o[c] = (_Float16)(gamma[c] * (xr[c] - mu) * inv + beta[c]);
}

// ---------------- WMMA GEMM: C[M,N] = A[M,K](f16) * Wsw[K,N](f16, swizzled) -
// 128 threads = 4 waves; block tile 64(M) x 64(N).
// Wave w owns N-tile n0+w*16 (its B-frag = one 32B global read/lane/k-step)
// and sweeps 4 M-subtiles via A-frags from LDS. 4 WMMAs/wave/k-step.
// mode 0: out = (acc+bias)*scale -> f16, [B,H,S,DH] head-split   (Q, K)
// mode 1: out = relu(acc+bias)   -> f16, row-major [M, DFF]      (FFN1)
// mode 2: out = acc+bias+resid   -> f32, row-major [M, D]        (FFN2)
// mode 3: out = acc+bias         -> f16, per-(b,h) B-frag-swizzled (V)

__global__ __launch_bounds__(128)
void gemm_wmma_kernel(const _Float16* __restrict__ A, int lda,
                      const _Float16* __restrict__ Wsw, int N,
                      const float* __restrict__ bias,
                      int K, int mode, float scale,
                      _Float16* __restrict__ outh,
                      float* __restrict__ outf,
                      const float* __restrict__ resid) {
    __shared__ _Float16 Alds[64 * 32];
    const int tid  = threadIdx.x;
    const int lane = tid & 31;
    const int wave = tid >> 5;
    const int m0   = blockIdx.y * 64;
    const int n0   = blockIdx.x * 64;

    const v8f zero = {0.f, 0.f, 0.f, 0.f, 0.f, 0.f, 0.f, 0.f};
    v8f acc[4];
#pragma unroll
    for (int mt = 0; mt < 4; ++mt) acc[mt] = zero;

    // Per-lane base into swizzled weights for this wave's N-tile.
    const int ntiles = N >> 4;
    const _Float16* wbase = Wsw + (((n0 >> 4) + wave) << 9) + (lane << 4);

    for (int k0 = 0; k0 < K; k0 += 32) {
        __syncthreads();
        {   // stage A tile 64x32 f16: 128 threads x 32B (2x uint4)
            const int idx = tid * 16;
            const int row = idx >> 5, col = idx & 31;
            const uint4* src =
                (const uint4*)(A + (size_t)(m0 + row) * lda + k0 + col);
            uint4* dst = (uint4*)(Alds + row * 32 + col);
            dst[0] = src[0];
            dst[1] = src[1];
        }
        __syncthreads();
        const v16h bw = load_v16h(wbase + ((size_t)(k0 >> 5) * ntiles << 9));
#pragma unroll
        for (int mt = 0; mt < 4; ++mt) {
            v16h a = load_a_frag(Alds + mt * 16 * 32, 32, lane);
            acc[mt] = wmma16x16x32(a, bw, acc[mt]);
        }
    }

    const int rhi  = (lane & 16) ? 8 : 0;
    const int gcol = n0 + wave * 16 + (lane & 15);
#pragma unroll
    for (int mt = 0; mt < 4; ++mt) {
#pragma unroll
        for (int r = 0; r < 8; ++r) {
            const int grow = m0 + mt * 16 + rhi + r;
            float v = acc[mt][r] + bias[gcol];
            if (mode == 0) {
                v *= scale;
                const int head = gcol >> 6, dh = gcol & 63;
                const int bb = grow >> 12, s = grow & 4095;
                outh[(((size_t)bb * H_ + head) * S_ + s) * DH_ + dh] = (_Float16)v;
            } else if (mode == 1) {
                outh[(size_t)grow * DFF_ + gcol] = (_Float16)fmaxf(v, 0.f);
            } else if (mode == 2) {
                outf[(size_t)grow * D_ + gcol] =
                    v + resid[(size_t)grow * D_ + gcol];
            } else { // mode 3: V, fragment-swizzled per (b,h) over [S,DH]
                const int head = gcol >> 6, dh = gcol & 63;
                const int bb = grow >> 12, s = grow & 4095;
                _Float16* vb = outh + ((size_t)bb * H_ + head) * (S_ * DH_);
                vb[bswz_idx(s, dh, DH_)] = (_Float16)v;
            }
        }
    }
}

// ---------------- Flash attention --------------------------------------------
// Block = 128 threads = 4 waves; each wave owns one 16-query tile of one (b,h).
// Q pre-scaled by 1/sqrt(DH). Keys streamed in chunks of 32:
//   scores 16x32: 4 WMMAs (K-frags = contiguous 32B vector loads);
//   online softmax (width-16 shuffles match C/D layout);
//   P->LDS->A-frag; P*V: 4 WMMAs with V-frags read directly from the
//   pre-swizzled V buffer (one 32B global read per frag per lane).

__global__ __launch_bounds__(128)
void attn_kernel(const _Float16* __restrict__ Q,    // [B,H,S,DH] (scaled)
                 const _Float16* __restrict__ Kh,   // [B,H,S,DH]
                 const _Float16* __restrict__ Vsw,  // per-(b,h) frag-swizzled
                 const float* __restrict__ mask,    // [B,S,S]
                 float* __restrict__ ctx) {         // [B,S,D] merged heads
    const int tid  = threadIdx.x;
    const int lane = tid & 31;
    const int wave = tid >> 5;
    const int blocks_per_bh = (S_ / 16) / 4;            // 64
    const int bh    = blockIdx.x / blocks_per_bh;
    const int qt    = (blockIdx.x % blocks_per_bh) * 4 + wave;
    const int b     = bh / H_;
    const int h     = bh % H_;
    const int qbase = qt * 16;

    const _Float16* qp = Q   + (size_t)bh * S_ * DH_;
    const _Float16* kp = Kh  + (size_t)bh * S_ * DH_;
    const _Float16* vp = Vsw + (size_t)bh * S_ * DH_;

    // Q A-fragments for contraction dh 0..31 and 32..63
    const int arow = lane & 15;
    const int ak0  = (lane & 16) ? 16 : 0;
    v16h aq0, aq1;
    {
        const _Float16* qr = qp + (size_t)(qbase + arow) * DH_;
#pragma unroll
        for (int j = 0; j < 8; ++j) { aq0[j] = qr[ak0 + j];      aq0[8 + j] = qr[ak0 + 8 + j]; }
#pragma unroll
        for (int j = 0; j < 8; ++j) { aq1[j] = qr[32 + ak0 + j]; aq1[8 + j] = qr[32 + ak0 + 8 + j]; }
    }

    __shared__ _Float16 Plds[4][16 * 32];   // per-wave P staging (D->A transpose)

    const v8f zero = {0.f, 0.f, 0.f, 0.f, 0.f, 0.f, 0.f, 0.f};
    float m_run[8], l_run[8];
    v8f o[4];
#pragma unroll
    for (int r = 0; r < 8; ++r) { m_run[r] = -1e30f; l_run[r] = 0.f; }
#pragma unroll
    for (int d = 0; d < 4; ++d) o[d] = zero;

    const int bn  = lane & 15;               // B-frag column
    const int bk0 = (lane & 16) ? 16 : 0;    // B-frag contraction base
    const int rhi = (lane & 16) ? 8 : 0;     // C/D row base

    for (int kc = 0; kc < S_; kc += 32) {
        // ---- scores: two 16x16 tiles (keys kc..+15, kc+16..+31) ----
        float p[2][8];
        float mc[8];
#pragma unroll
        for (int r = 0; r < 8; ++r) mc[r] = -1e30f;
#pragma unroll
        for (int t = 0; t < 2; ++t) {
            const int key = kc + t * 16 + bn;
            const _Float16* kr = kp + (size_t)key * DH_;
            const v16h bk_lo = load_v16h(kr + bk0);       // dh 0..31 slice
            const v16h bk_hi = load_v16h(kr + 32 + bk0);  // dh 32..63 slice
            v8f s = zero;
            s = wmma16x16x32(aq0, bk_lo, s);
            s = wmma16x16x32(aq1, bk_hi, s);
            const float* mrow = mask + ((size_t)b * S_ + qbase) * S_;
#pragma unroll
            for (int r = 0; r < 8; ++r) {
                const float sv = s[r] * mrow[(size_t)(rhi + r) * S_ + key];
                p[t][r] = sv;
                mc[r] = fmaxf(mc[r], sv);
            }
        }
        // row-max across the 16 lanes sharing each row (width-16 groups)
#pragma unroll
        for (int r = 0; r < 8; ++r)
#pragma unroll
            for (int off = 1; off < 16; off <<= 1)
                mc[r] = fmaxf(mc[r], __shfl_xor(mc[r], off, 16));

        // ---- online softmax update ----
#pragma unroll
        for (int r = 0; r < 8; ++r) {
            const float m_new = fmaxf(m_run[r], mc[r]);
            const float alpha = __expf(m_run[r] - m_new);
            m_run[r] = m_new;
            const float p0 = __expf(p[0][r] - m_new);
            const float p1 = __expf(p[1][r] - m_new);
            p[0][r] = p0; p[1][r] = p1;
            float rsum = p0 + p1;
#pragma unroll
            for (int off = 1; off < 16; off <<= 1) rsum += __shfl_xor(rsum, off, 16);
            l_run[r] = l_run[r] * alpha + rsum;
#pragma unroll
            for (int d = 0; d < 4; ++d) o[d][r] = o[d][r] * alpha;
        }

        // ---- P (16x32) to LDS, reshape to A-fragment ----
        {
            _Float16* pl = Plds[wave];
#pragma unroll
            for (int t = 0; t < 2; ++t)
#pragma unroll
                for (int r = 0; r < 8; ++r)
                    pl[(rhi + r) * 32 + t * 16 + bn] = (_Float16)p[t][r];
        }
        __syncthreads();

        // ---- O += P * Vchunk: V-frags straight from swizzled global ----
        const v16h ap = load_a_frag(Plds[wave], 32, lane);
        const _Float16* vchunk = vp + (((size_t)(kc >> 5) * 4) << 9) + (lane << 4);
#pragma unroll
        for (int d = 0; d < 4; ++d) {
            const v16h bv = load_v16h(vchunk + (d << 9));
            o[d] = wmma16x16x32(ap, bv, o[d]);
        }
    }

    // ---- normalize and write merged-head ctx ----
#pragma unroll
    for (int d = 0; d < 4; ++d) {
        const int dh = d * 16 + bn;
#pragma unroll
        for (int r = 0; r < 8; ++r) {
            const int row = qbase + rhi + r;
            ctx[((size_t)b * S_ + row) * D_ + h * DH_ + dh] = o[d][r] / l_run[r];
        }
    }
}

// ---------------- elementwise residual add ----------------------------------

__global__ __launch_bounds__(256)
void add_kernel(const float* __restrict__ a, const float* __restrict__ b,
                float* __restrict__ out, int n) {
    const int i = blockIdx.x * 256 + threadIdx.x;
    if (i < n) out[i] = a[i] + b[i];
}

// ---------------- launch ------------------------------------------------------

extern "C" void kernel_launch(void* const* d_in, const int* in_sizes, int n_in,
                              void* d_out, int out_size, void* d_ws, size_t ws_size,
                              hipStream_t stream) {
    const float* x      = (const float*)d_in[0];
    const float* mask   = (const float*)d_in[1];
    const float* wq     = (const float*)d_in[2];
    const float* bq     = (const float*)d_in[3];
    const float* wk     = (const float*)d_in[4];
    const float* bk     = (const float*)d_in[5];
    const float* wv     = (const float*)d_in[6];
    const float* bv     = (const float*)d_in[7];
    // d_in[8]=wo, d_in[9]=bo: intentionally unused (reference skips out_linear)
    const float* w1     = (const float*)d_in[10];
    const float* b1     = (const float*)d_in[11];
    const float* w2     = (const float*)d_in[12];
    const float* b2     = (const float*)d_in[13];
    const float* gamma1 = (const float*)d_in[14];
    const float* beta1  = (const float*)d_in[15];
    const float* gamma2 = (const float*)d_in[16];
    const float* beta2  = (const float*)d_in[17];
    float* out = (float*)d_out;

    char* ws = (char*)d_ws;
    size_t off = 0;
    auto walloc = [&](size_t bytes) -> void* {
        void* p = ws + off;
        off += (bytes + 255) & ~(size_t)255;
        return p;
    };
    _Float16* l_h   = (_Float16*)walloc((size_t)NROW * D_   * 2);
    _Float16* q_h   = (_Float16*)walloc((size_t)NROW * D_   * 2);
    _Float16* k_h   = (_Float16*)walloc((size_t)NROW * D_   * 2);
    _Float16* v_sw  = (_Float16*)walloc((size_t)NROW * D_   * 2);
    float*    ctx   = (float*)   walloc((size_t)NROW * D_   * 4);
    float*    x1    = (float*)   walloc((size_t)NROW * D_   * 4);
    _Float16* f1_h  = (_Float16*)walloc((size_t)NROW * DFF_ * 2);
    _Float16* wq_sw = (_Float16*)walloc((size_t)D_ * D_   * 2);
    _Float16* wk_sw = (_Float16*)walloc((size_t)D_ * D_   * 2);
    _Float16* wv_sw = (_Float16*)walloc((size_t)D_ * D_   * 2);
    _Float16* w1_sw = (_Float16*)walloc((size_t)D_ * DFF_ * 2);
    _Float16* w2_sw = (_Float16*)walloc((size_t)DFF_ * D_ * 2);
    (void)in_sizes; (void)n_in; (void)out_size; (void)ws_size;

    // 0) one-time weight convert + fragment swizzle (fp32 -> f16)
    wswz_kernel<<<(D_ * D_ + 255) / 256, 256, 0, stream>>>(wq, wq_sw, D_, D_);
    wswz_kernel<<<(D_ * D_ + 255) / 256, 256, 0, stream>>>(wk, wk_sw, D_, D_);
    wswz_kernel<<<(D_ * D_ + 255) / 256, 256, 0, stream>>>(wv, wv_sw, D_, D_);
    wswz_kernel<<<(D_ * DFF_ + 255) / 256, 256, 0, stream>>>(w1, w1_sw, D_, DFF_);
    wswz_kernel<<<(DFF_ * D_ + 255) / 256, 256, 0, stream>>>(w2, w2_sw, DFF_, D_);

    const float qscale = 0.125f; // 1/sqrt(DH)

    // 1) LN1 -> l1 (f16)
    ln_kernel<<<NROW, 256, 0, stream>>>(x, gamma1, beta1, l_h);

    // 2) QKV projections; Q scaled; V stored fragment-swizzled for attention
    dim3 gqkv(D_ / 64, NROW / 64);          // (12, 128)
    gemm_wmma_kernel<<<gqkv, 128, 0, stream>>>(l_h, D_, wq_sw, D_, bq, D_, 0,
                                               qscale, q_h, nullptr, nullptr);
    gemm_wmma_kernel<<<gqkv, 128, 0, stream>>>(l_h, D_, wk_sw, D_, bk, D_, 0,
                                               1.0f, k_h, nullptr, nullptr);
    gemm_wmma_kernel<<<gqkv, 128, 0, stream>>>(l_h, D_, wv_sw, D_, bv, D_, 3,
                                               1.0f, v_sw, nullptr, nullptr);

    // 3) flash attention -> ctx (fp32, merged heads)
    attn_kernel<<<B_ * H_ * ((S_ / 16) / 4), 128, 0, stream>>>(q_h, k_h, v_sw,
                                                               mask, ctx);

    // 4) x1 = x + ctx
    add_kernel<<<(NROW * D_ + 255) / 256, 256, 0, stream>>>(x, ctx, x1, NROW * D_);

    // 5) LN2 -> l2 (f16, reuse buffer)
    ln_kernel<<<NROW, 256, 0, stream>>>(x1, gamma2, beta2, l_h);

    // 6) FFN1: relu(l2 @ w1 + b1) -> f16
    dim3 gff1(DFF_ / 64, NROW / 64);        // (48, 128)
    gemm_wmma_kernel<<<gff1, 128, 0, stream>>>(l_h, D_, w1_sw, DFF_, b1, D_, 1,
                                               1.0f, f1_h, nullptr, nullptr);

    // 7) FFN2: f1 @ w2 + b2 + x1 -> fp32 output
    dim3 gff2(D_ / 64, NROW / 64);          // (12, 128)
    gemm_wmma_kernel<<<gff2, 128, 0, stream>>>(f1_h, DFF_, w2_sw, D_, b2, DFF_, 2,
                                               1.0f, nullptr, out, x1);
}
